// ConFNet7_90812788506966
// MI455X (gfx1250) — compile-verified, matrix operand order checked by reading, not run on hardware
//
#include <hip/hip_runtime.h>
#include <hip/hip_bf16.h>
#include <math.h>

typedef __attribute__((ext_vector_type(16))) _Float16 v16h;
typedef __attribute__((ext_vector_type(8)))  _Float16 v8h;
typedef __attribute__((ext_vector_type(8)))  float    v8f;

#define EPS 1e-5f
#define NB 32768

__device__ __forceinline__ float sigf(float x)  { return 1.0f / (1.0f + __expf(-x)); }
__device__ __forceinline__ float lrelu(float x) { return x > 0.0f ? x : 0.01f * x; }

// ---------------------------------------------------------------- utilities
__global__ void k_zero(float* p, int n) {
    int i = blockIdx.x * blockDim.x + threadIdx.x;
    if (i < n) p[i] = 0.0f;
}

__global__ void k_f32_to_f16(const float* __restrict__ src, _Float16* __restrict__ dst, int n) {
    int i = blockIdx.x * blockDim.x + threadIdx.x;
    if (i < n) dst[i] = (_Float16)src[i];
}

// finalize BN: scale = g/sqrt(var+eps), shift = b - mean*scale
__global__ void k_finalize(const float* __restrict__ sum, const float* __restrict__ sumsq,
                           const float* __restrict__ g, const float* __restrict__ bt,
                           float* __restrict__ scale, float* __restrict__ shift,
                           int n, float count) {
    int i = threadIdx.x;
    if (i < n) {
        float m  = sum[i] / count;
        float v  = sumsq[i] / count - m * m;
        float sc = g[i] * rsqrtf(v + EPS);
        scale[i] = sc;
        shift[i] = bt[i] - m * sc;
    }
}

// per-column (feature) stats of an NB x N matrix, LDS-accumulated
__global__ void k_colstats(const float* __restrict__ Y, int N,
                           float* __restrict__ sum, float* __restrict__ sumsq) {
    __shared__ float ssum[128], ssq[128];
    for (int i = threadIdx.x; i < N; i += blockDim.x) { ssum[i] = 0.0f; ssq[i] = 0.0f; }
    __syncthreads();
    size_t total = (size_t)NB * (size_t)N;
    for (size_t idx = (size_t)blockIdx.x * blockDim.x + threadIdx.x; idx < total;
         idx += (size_t)gridDim.x * blockDim.x) {
        float v = Y[idx];
        int col = (int)(idx % (size_t)N);
        atomicAdd(&ssum[col], v);
        atomicAdd(&ssq[col],  v * v);
    }
    __syncthreads();
    for (int i = threadIdx.x; i < N; i += blockDim.x) {
        atomicAdd(&sum[i],   ssum[i]);
        atomicAdd(&sumsq[i], ssq[i]);
    }
}

// ---------------------------------------------------------------- conv stats
// conv1 raw output per (b, o, l):  l=0,3 -> cb1[o];  l=1,2 -> cw1[o]*x[l-1]+cb1[o]
__global__ void k_conv1_stats(const float* __restrict__ x, const float* __restrict__ cw1,
                              const float* __restrict__ cb1,
                              float* __restrict__ sum, float* __restrict__ sumsq) {
    __shared__ float ssum[8], ssq[8];
    if (threadIdx.x < 8) { ssum[threadIdx.x] = 0.0f; ssq[threadIdx.x] = 0.0f; }
    __syncthreads();
    int b = blockIdx.x * blockDim.x + threadIdx.x;
    if (b < NB) {
        float x0 = x[2 * b], x1 = x[2 * b + 1];
#pragma unroll
        for (int o = 0; o < 8; ++o) {
            float w = cw1[o], c = cb1[o];
            float v1 = w * x0 + c, v2 = w * x1 + c;
            atomicAdd(&ssum[o], 2.0f * c + v1 + v2);
            atomicAdd(&ssq[o],  2.0f * c * c + v1 * v1 + v2 * v2);
        }
    }
    __syncthreads();
    if (threadIdx.x < 8) {
        atomicAdd(&sum[threadIdx.x],   ssum[threadIdx.x]);
        atomicAdd(&sumsq[threadIdx.x], ssq[threadIdx.x]);
    }
}

__device__ __forceinline__ void compute_a1(float x0, float x1,
                                           const float* cw1, const float* cb1,
                                           const float* c1s, const float* c1t,
                                           float a1[8][4]) {
#pragma unroll
    for (int o = 0; o < 8; ++o) {
        float w = cw1[o], cb = cb1[o], sc = c1s[o], sh = c1t[o];
        float e = lrelu(cb * sc + sh);
        a1[o][0] = e;
        a1[o][1] = lrelu((w * x0 + cb) * sc + sh);
        a1[o][2] = lrelu((w * x1 + cb) * sc + sh);
        a1[o][3] = e;
    }
}

// conv2 raw stats: positions 0,5 are conv over zero-pad -> cb2[t]
__global__ void k_conv2_stats(const float* __restrict__ x, const float* __restrict__ cw1,
                              const float* __restrict__ cb1,
                              const float* __restrict__ c1s, const float* __restrict__ c1t,
                              const float* __restrict__ cw2, const float* __restrict__ cb2,
                              float* __restrict__ sum, float* __restrict__ sumsq) {
    __shared__ float ssum[64], ssq[64], sw2[512], sb2[64];
    for (int i = threadIdx.x; i < 512; i += blockDim.x) sw2[i] = cw2[i];
    for (int i = threadIdx.x; i < 64;  i += blockDim.x) { sb2[i] = cb2[i]; ssum[i] = 0.0f; ssq[i] = 0.0f; }
    __syncthreads();
    int b = blockIdx.x * blockDim.x + threadIdx.x;
    if (b < NB) {
        float a1[8][4];
        compute_a1(x[2 * b], x[2 * b + 1], cw1, cb1, c1s, c1t, a1);
#pragma unroll 1
        for (int t = 0; t < 64; ++t) {
            float cb = sb2[t];
            float s = 2.0f * cb, sq = 2.0f * cb * cb;
#pragma unroll
            for (int l = 0; l < 4; ++l) {
                float acc = cb;
#pragma unroll
                for (int i = 0; i < 8; ++i) acc += sw2[t * 8 + i] * a1[i][l];
                s += acc; sq += acc * acc;
            }
            atomicAdd(&ssum[t], s);
            atomicAdd(&ssq[t],  sq);
        }
    }
    __syncthreads();
    for (int i = threadIdx.x; i < 64; i += blockDim.x) {
        atomicAdd(&sum[i],   ssum[i]);
        atomicAdd(&sumsq[i], ssq[i]);
    }
}

// ---------------------------------------------------------------- front + 5 LSTM layers
template <int NIN>
__device__ __forceinline__ void lstm_step(const float* __restrict__ Wih,
                                          const float* __restrict__ Whh,
                                          const float* __restrict__ bsum,
                                          const float* __restrict__ xin,
                                          float* __restrict__ h, float* __restrict__ c) {
    float z[20];
#pragma unroll
    for (int g = 0; g < 20; ++g) {
        float a = bsum[g];
#pragma unroll
        for (int j = 0; j < NIN; ++j) a += Wih[g * NIN + j] * xin[j];
#pragma unroll
        for (int j = 0; j < 5; ++j) a += Whh[g * 5 + j] * h[j];
        z[g] = a;
    }
    float hn[5];
#pragma unroll
    for (int j = 0; j < 5; ++j) {
        float cn = sigf(z[5 + j]) * c[j] + sigf(z[j]) * tanhf(z[10 + j]);
        c[j]  = cn;
        hn[j] = sigf(z[15 + j]) * tanhf(cn);
    }
#pragma unroll
    for (int j = 0; j < 5; ++j) h[j] = hn[j];
}

__global__ void k_front_lstm(const float* __restrict__ x,
                             const float* __restrict__ cw1, const float* __restrict__ cb1,
                             const float* __restrict__ c1s, const float* __restrict__ c1t,
                             const float* __restrict__ cw2, const float* __restrict__ cb2,
                             const float* __restrict__ c2s, const float* __restrict__ c2t,
                             const float* __restrict__ Wih0, const float* __restrict__ Wih,
                             const float* __restrict__ Whh,
                             const float* __restrict__ bih, const float* __restrict__ bhh,
                             _Float16* __restrict__ f) {
    __shared__ float sW0[120], sWi[400], sWh[500], sB[100];
    __shared__ float sw2[512], scb2[64], s2s[64], s2t[64];
    __shared__ float scw1[8], scb1[8], s1s[8], s1t[8];
    for (int i = threadIdx.x; i < 512; i += blockDim.x) sw2[i] = cw2[i];
    for (int i = threadIdx.x; i < 500; i += blockDim.x) sWh[i] = Whh[i];
    for (int i = threadIdx.x; i < 400; i += blockDim.x) sWi[i] = Wih[i];
    for (int i = threadIdx.x; i < 120; i += blockDim.x) sW0[i] = Wih0[i];
    for (int i = threadIdx.x; i < 100; i += blockDim.x) sB[i] = bih[i] + bhh[i];
    for (int i = threadIdx.x; i < 64;  i += blockDim.x) { scb2[i] = cb2[i]; s2s[i] = c2s[i]; s2t[i] = c2t[i]; }
    if (threadIdx.x < 8) {
        scw1[threadIdx.x] = cw1[threadIdx.x]; scb1[threadIdx.x] = cb1[threadIdx.x];
        s1s[threadIdx.x]  = c1s[threadIdx.x]; s1t[threadIdx.x]  = c1t[threadIdx.x];
    }
    __syncthreads();
    int b = blockIdx.x * blockDim.x + threadIdx.x;
    if (b >= NB) return;

    float a1[8][4];
    compute_a1(x[2 * b], x[2 * b + 1], scw1, scb1, s1s, s1t, a1);

    float h0[5], c0[5], h1[5], c1[5], h2[5], c2[5], h3[5], c3[5], h4[5], c4[5];
#pragma unroll
    for (int j = 0; j < 5; ++j) {
        h0[j] = c0[j] = h1[j] = c1[j] = h2[j] = c2[j] = h3[j] = c3[j] = h4[j] = c4[j] = 0.0f;
    }

#pragma unroll 1
    for (int t = 0; t < 64; ++t) {
        // conv2 + bn2 + lrelu, time slice t (6 features)
        float xt[6];
        {
            float cb = scb2[t], sc = s2s[t], sh = s2t[t];
            float e = lrelu(cb * sc + sh);
            xt[0] = e; xt[5] = e;
#pragma unroll
            for (int l = 0; l < 4; ++l) {
                float acc = cb;
#pragma unroll
                for (int i = 0; i < 8; ++i) acc += sw2[t * 8 + i] * a1[i][l];
                xt[l + 1] = lrelu(acc * sc + sh);
            }
        }
        lstm_step<6>(sW0,       sWh,       sB,      xt, h0, c0);
        lstm_step<5>(sWi,       sWh + 100, sB + 20, h0, h1, c1);
        lstm_step<5>(sWi + 100, sWh + 200, sB + 40, h1, h2, c2);
        lstm_step<5>(sWi + 200, sWh + 300, sB + 60, h2, h3, c3);
        lstm_step<5>(sWi + 300, sWh + 400, sB + 80, h3, h4, c4);
        size_t base = (size_t)b * 320 + t * 5;
#pragma unroll
        for (int j = 0; j < 5; ++j) f[base + j] = (_Float16)h4[j];
    }
}

// ---------------------------------------------------------------- WMMA GEMMs
// GEMM1: Y1(NB x 128) = f(NB x 320, f16) @ w1h^T(128 x 320, f16) + b1, f32 out.
// One wave per 16x16 tile; A-layout: lane L -> row L&15, halves j -> K=(j<8?0:16)+(L>>4)*8+(j&7).
// Fully unrolled, software-pipelined: iteration k+1 loads are issued before iteration k's WMMA.
__global__ void k_gemm1_wmma(const _Float16* __restrict__ A, const _Float16* __restrict__ Bw,
                             const float* __restrict__ bias, float* __restrict__ Y) {
    int lane  = threadIdx.x & 31;
    int wave  = threadIdx.x >> 5;
    int mbase = (blockIdx.x * 8 + wave) * 16;
    int nbase = blockIdx.y * 16;
    int mr    = lane & 15;
    int q     = lane >> 4;
    int ncol  = nbase + mr;

    v8f acc;
    float bv = bias[ncol];
#pragma unroll
    for (int r = 0; r < 8; ++r) acc[r] = bv;

    const _Float16* arow = A  + (size_t)(mbase + mr) * 320 + q * 8;
    const _Float16* brow = Bw + (size_t)ncol * 320 + q * 8;

    // prologue: iteration-0 operands in flight
    v8h alo = *(const v8h*)(arow);
    v8h ahi = *(const v8h*)(arow + 16);
    v8h blo = *(const v8h*)(brow);
    v8h bhi = *(const v8h*)(brow + 16);

#pragma unroll
    for (int k0 = 0; k0 < 320; k0 += 32) {
        v16h a, bmat;
#pragma unroll
        for (int j = 0; j < 8; ++j) {
            a[j] = alo[j];    a[j + 8] = ahi[j];
            bmat[j] = blo[j]; bmat[j + 8] = bhi[j];
        }
        if (k0 + 32 < 320) {  // issue next iteration's loads before this WMMA
            alo = *(const v8h*)(arow + k0 + 32);
            ahi = *(const v8h*)(arow + k0 + 48);
            blo = *(const v8h*)(brow + k0 + 32);
            bhi = *(const v8h*)(brow + k0 + 48);
        }
        acc = __builtin_amdgcn_wmma_f32_16x16x32_f16(false, a, false, bmat,
                                                     (short)0, acc, false, false);
    }
#pragma unroll
    for (int r = 0; r < 8; ++r) {
        int row = mbase + q * 8 + r;
        Y[(size_t)row * 128 + ncol] = acc[r];
    }
}

// GEMM2: Y2(NB x 64) = tanh(bn1(Y1))(f16, built on the fly) @ w2h^T(64 x 128) + b2.
// Fully unrolled (4 K-steps) so all B loads can be hoisted; A operand is VALU-generated.
__global__ void k_gemm2_wmma(const float* __restrict__ Y1,
                             const float* __restrict__ s1, const float* __restrict__ t1,
                             const _Float16* __restrict__ Bw, const float* __restrict__ bias,
                             float* __restrict__ Y2) {
    __shared__ float ss[128], sh[128];
    for (int i = threadIdx.x; i < 128; i += blockDim.x) { ss[i] = s1[i]; sh[i] = t1[i]; }
    __syncthreads();

    int lane  = threadIdx.x & 31;
    int wave  = threadIdx.x >> 5;
    int mbase = (blockIdx.x * 8 + wave) * 16;
    int nbase = blockIdx.y * 16;
    int mr    = lane & 15;
    int q     = lane >> 4;
    int ncol  = nbase + mr;

    v8f acc;
    float bv = bias[ncol];
#pragma unroll
    for (int r = 0; r < 8; ++r) acc[r] = bv;

    const float*    arow = Y1 + (size_t)(mbase + mr) * 128;
    const _Float16* brow = Bw + (size_t)ncol * 128 + q * 8;
#pragma unroll
    for (int k0 = 0; k0 < 128; k0 += 32) {
        v8h blo = *(const v8h*)(brow + k0);
        v8h bhi = *(const v8h*)(brow + k0 + 16);
        v16h a, bmat;
#pragma unroll
        for (int j = 0; j < 16; ++j) {
            int k = k0 + ((j < 8) ? 0 : 16) + q * 8 + (j & 7);
            a[j] = (_Float16)tanhf(arow[k] * ss[k] + sh[k]);
        }
#pragma unroll
        for (int j = 0; j < 8; ++j) { bmat[j] = blo[j]; bmat[j + 8] = bhi[j]; }
        acc = __builtin_amdgcn_wmma_f32_16x16x32_f16(false, a, false, bmat,
                                                     (short)0, acc, false, false);
    }
#pragma unroll
    for (int r = 0; r < 8; ++r) {
        int row = mbase + q * 8 + r;
        Y2[(size_t)row * 64 + ncol] = acc[r];
    }
}

// ---------------------------------------------------------------- tail GEMVs with fused stats
// x2 = tanh(bn2(Y2)); Y3 = x2 @ w3^T + b3 (64 -> 4), + col stats of Y3
__global__ void k_l3(const float* __restrict__ Y2, const float* __restrict__ s2,
                     const float* __restrict__ t2, const float* __restrict__ w3,
                     const float* __restrict__ b3, float* __restrict__ Y3,
                     float* __restrict__ sum, float* __restrict__ sumsq) {
    __shared__ float sw[256], ss[64], sh[64], sb[4], ssum[4], ssq[4];
    for (int i = threadIdx.x; i < 256; i += blockDim.x) sw[i] = w3[i];
    for (int i = threadIdx.x; i < 64;  i += blockDim.x) { ss[i] = s2[i]; sh[i] = t2[i]; }
    if (threadIdx.x < 4) { sb[threadIdx.x] = b3[threadIdx.x]; ssum[threadIdx.x] = 0.0f; ssq[threadIdx.x] = 0.0f; }
    __syncthreads();
    int b = blockIdx.x * blockDim.x + threadIdx.x;
    if (b < NB) {
        float x2[64];
#pragma unroll
        for (int i = 0; i < 64; ++i) x2[i] = tanhf(Y2[(size_t)b * 64 + i] * ss[i] + sh[i]);
#pragma unroll
        for (int j = 0; j < 4; ++j) {
            float a = sb[j];
#pragma unroll
            for (int i = 0; i < 64; ++i) a += sw[j * 64 + i] * x2[i];
            Y3[(size_t)b * 4 + j] = a;
            atomicAdd(&ssum[j], a);
            atomicAdd(&ssq[j],  a * a);
        }
    }
    __syncthreads();
    if (threadIdx.x < 4) {
        atomicAdd(&sum[threadIdx.x],   ssum[threadIdx.x]);
        atomicAdd(&sumsq[threadIdx.x], ssq[threadIdx.x]);
    }
}

// x3 = tanh(bn3(Y3)); Y4 = x3 @ w4^T + b4 (4 -> 32), + stats
__global__ void k_l4(const float* __restrict__ Y3, const float* __restrict__ s3,
                     const float* __restrict__ t3, const float* __restrict__ w4,
                     const float* __restrict__ b4, float* __restrict__ Y4,
                     float* __restrict__ sum, float* __restrict__ sumsq) {
    __shared__ float sw[128], sb[32], ss[4], sh[4], ssum[32], ssq[32];
    for (int i = threadIdx.x; i < 128; i += blockDim.x) sw[i] = w4[i];
    for (int i = threadIdx.x; i < 32;  i += blockDim.x) { sb[i] = b4[i]; ssum[i] = 0.0f; ssq[i] = 0.0f; }
    if (threadIdx.x < 4) { ss[threadIdx.x] = s3[threadIdx.x]; sh[threadIdx.x] = t3[threadIdx.x]; }
    __syncthreads();
    int b = blockIdx.x * blockDim.x + threadIdx.x;
    if (b < NB) {
        float x3[4];
#pragma unroll
        for (int j = 0; j < 4; ++j) x3[j] = tanhf(Y3[(size_t)b * 4 + j] * ss[j] + sh[j]);
#pragma unroll
        for (int o = 0; o < 32; ++o) {
            float a = sb[o];
#pragma unroll
            for (int j = 0; j < 4; ++j) a += sw[o * 4 + j] * x3[j];
            Y4[(size_t)b * 32 + o] = a;
            atomicAdd(&ssum[o], a);
            atomicAdd(&ssq[o],  a * a);
        }
    }
    __syncthreads();
    for (int i = threadIdx.x; i < 32; i += blockDim.x) {
        atomicAdd(&sum[i],   ssum[i]);
        atomicAdd(&sumsq[i], ssq[i]);
    }
}

// x4 = tanh(bn4(Y4)); Y5 = x4 @ w5^T + b5 (32 -> 32), + stats
__global__ void k_l5(const float* __restrict__ Y4, const float* __restrict__ s4,
                     const float* __restrict__ t4, const float* __restrict__ w5,
                     const float* __restrict__ b5, float* __restrict__ Y5,
                     float* __restrict__ sum, float* __restrict__ sumsq) {
    __shared__ float sw[1024], sb[32], ss[32], sh[32], ssum[32], ssq[32];
    for (int i = threadIdx.x; i < 1024; i += blockDim.x) sw[i] = w5[i];
    for (int i = threadIdx.x; i < 32; i += blockDim.x) {
        sb[i] = b5[i]; ss[i] = s4[i]; sh[i] = t4[i]; ssum[i] = 0.0f; ssq[i] = 0.0f;
    }
    __syncthreads();
    int b = blockIdx.x * blockDim.x + threadIdx.x;
    if (b < NB) {
        float x4[32];
#pragma unroll
        for (int i = 0; i < 32; ++i) x4[i] = tanhf(Y4[(size_t)b * 32 + i] * ss[i] + sh[i]);
#pragma unroll
        for (int o = 0; o < 32; ++o) {
            float a = sb[o];
#pragma unroll
            for (int i = 0; i < 32; ++i) a += sw[o * 32 + i] * x4[i];
            Y5[(size_t)b * 32 + o] = a;
            atomicAdd(&ssum[o], a);
            atomicAdd(&ssq[o],  a * a);
        }
    }
    __syncthreads();
    for (int i = threadIdx.x; i < 32; i += blockDim.x) {
        atomicAdd(&sum[i],   ssum[i]);
        atomicAdd(&sumsq[i], ssq[i]);
    }
}

// x5 = x4 + tanh(bn5(Y5)); x6 = tanh(x5@w6^T+b6)+x3; x7 = tanh(x6@w7^T+b7);
// out = sigmoid(x7@w8^T+b8)
__global__ void k_final(const float* __restrict__ Y3, const float* __restrict__ s3, const float* __restrict__ t3,
                        const float* __restrict__ Y4, const float* __restrict__ s4, const float* __restrict__ t4,
                        const float* __restrict__ Y5, const float* __restrict__ s5, const float* __restrict__ t5,
                        const float* __restrict__ w6, const float* __restrict__ b6,
                        const float* __restrict__ w7, const float* __restrict__ b7,
                        const float* __restrict__ w8, const float* __restrict__ b8,
                        float* __restrict__ out) {
    __shared__ float sw6[128], sw7[2048], sw8[2048], sb7[512];
    __shared__ float sb6[4], sb8[4], a3s[4], a3t[4], a4s[32], a4t[32], a5s[32], a5t[32];
    for (int i = threadIdx.x; i < 2048; i += blockDim.x) { sw7[i] = w7[i]; sw8[i] = w8[i]; }
    for (int i = threadIdx.x; i < 512;  i += blockDim.x) sb7[i] = b7[i];
    for (int i = threadIdx.x; i < 128;  i += blockDim.x) sw6[i] = w6[i];
    for (int i = threadIdx.x; i < 32;   i += blockDim.x) { a4s[i] = s4[i]; a4t[i] = t4[i]; a5s[i] = s5[i]; a5t[i] = t5[i]; }
    if (threadIdx.x < 4) {
        sb6[threadIdx.x] = b6[threadIdx.x]; sb8[threadIdx.x] = b8[threadIdx.x];
        a3s[threadIdx.x] = s3[threadIdx.x]; a3t[threadIdx.x] = t3[threadIdx.x];
    }
    __syncthreads();
    int b = blockIdx.x * blockDim.x + threadIdx.x;
    if (b >= NB) return;

    float x3[4];
#pragma unroll
    for (int j = 0; j < 4; ++j) x3[j] = tanhf(Y3[(size_t)b * 4 + j] * a3s[j] + a3t[j]);
    float x5[32];
#pragma unroll
    for (int i = 0; i < 32; ++i) {
        float x4 = tanhf(Y4[(size_t)b * 32 + i] * a4s[i] + a4t[i]);
        x5[i] = x4 + tanhf(Y5[(size_t)b * 32 + i] * a5s[i] + a5t[i]);
    }
    float x6[4];
#pragma unroll
    for (int j = 0; j < 4; ++j) {
        float a = sb6[j];
#pragma unroll
        for (int i = 0; i < 32; ++i) a += sw6[j * 32 + i] * x5[i];
        x6[j] = tanhf(a) + x3[j];
    }
    float acc0 = sb8[0], acc1 = sb8[1], acc2 = sb8[2], acc3 = sb8[3];
#pragma unroll 4
    for (int k = 0; k < 512; ++k) {
        float x7 = tanhf(sb7[k] + sw7[k * 4 + 0] * x6[0] + sw7[k * 4 + 1] * x6[1]
                                + sw7[k * 4 + 2] * x6[2] + sw7[k * 4 + 3] * x6[3]);
        acc0 += sw8[0 * 512 + k] * x7;
        acc1 += sw8[1 * 512 + k] * x7;
        acc2 += sw8[2 * 512 + k] * x7;
        acc3 += sw8[3 * 512 + k] * x7;
    }
    out[(size_t)b * 4 + 0] = sigf(acc0);
    out[(size_t)b * 4 + 1] = sigf(acc1);
    out[(size_t)b * 4 + 2] = sigf(acc2);
    out[(size_t)b * 4 + 3] = sigf(acc3);
}

// ---------------------------------------------------------------- launch
extern "C" void kernel_launch(void* const* d_in, const int* in_sizes, int n_in,
                              void* d_out, int out_size, void* d_ws, size_t ws_size,
                              hipStream_t stream) {
    const float* x    = (const float*)d_in[0];
    const float* cw1  = (const float*)d_in[1];
    const float* cb1  = (const float*)d_in[2];
    const float* gc1  = (const float*)d_in[3];
    const float* btc1 = (const float*)d_in[4];
    const float* cw2  = (const float*)d_in[5];
    const float* cb2  = (const float*)d_in[6];
    const float* gc2  = (const float*)d_in[7];
    const float* btc2 = (const float*)d_in[8];
    const float* Wih0 = (const float*)d_in[9];
    const float* Wih  = (const float*)d_in[10];
    const float* Whh  = (const float*)d_in[11];
    const float* bih  = (const float*)d_in[12];
    const float* bhh  = (const float*)d_in[13];
    const float* w1   = (const float*)d_in[14];
    const float* b1   = (const float*)d_in[15];
    const float* g1   = (const float*)d_in[16];
    const float* bt1  = (const float*)d_in[17];
    const float* w2   = (const float*)d_in[18];
    const float* b2   = (const float*)d_in[19];
    const float* g2   = (const float*)d_in[20];
    const float* bt2  = (const float*)d_in[21];
    const float* w3   = (const float*)d_in[22];
    const float* b3   = (const float*)d_in[23];
    const float* g3   = (const float*)d_in[24];
    const float* bt3  = (const float*)d_in[25];
    const float* w4   = (const float*)d_in[26];
    const float* b4   = (const float*)d_in[27];
    const float* g4   = (const float*)d_in[28];
    const float* bt4  = (const float*)d_in[29];
    const float* w5   = (const float*)d_in[30];
    const float* b5   = (const float*)d_in[31];
    const float* g5   = (const float*)d_in[32];
    const float* bt5  = (const float*)d_in[33];
    const float* w6   = (const float*)d_in[34];
    const float* b6   = (const float*)d_in[35];
    const float* w7   = (const float*)d_in[36];
    const float* b7   = (const float*)d_in[37];
    const float* w8   = (const float*)d_in[38];
    const float* b8   = (const float*)d_in[39];
    float* out = (float*)d_out;

    char* wsp = (char*)d_ws;
    size_t off = 0;
    auto take = [&](size_t bytes) -> char* {
        char* p = wsp + off;
        off += (bytes + 255) & ~(size_t)255;
        return p;
    };

    float*    st   = (float*)take(2048 * sizeof(float));   // all BN stats/params
    _Float16* w1h  = (_Float16*)take((size_t)40960 * 2);
    _Float16* w2h  = (_Float16*)take((size_t)8192 * 2);
    _Float16* fbuf = (_Float16*)take((size_t)NB * 320 * 2);
    float*    y1   = (float*)take((size_t)NB * 128 * 4);
    float*    y2   = (float*)take((size_t)NB * 64 * 4);
    float*    y3   = (float*)take((size_t)NB * 4 * 4);
    float*    y4   = (float*)take((size_t)NB * 32 * 4);
    float*    y5   = (float*)take((size_t)NB * 32 * 4);

    // stats layout inside st (floats):
    // c1: sum@0 sq@8 sc@16 sh@24 | c2: sum@32 sq@96 sc@160 sh@224
    // l1: sum@288 sq@416 sc@544 sh@672 | l2: sum@800 sq@864 sc@928 sh@992
    // l3: sum@1056 sq@1060 sc@1064 sh@1068 | l4: sum@1072 sq@1104 sc@1136 sh@1168
    // l5: sum@1200 sq@1232 sc@1264 sh@1296
    const float fB = (float)NB;

    k_zero<<<8, 256, 0, stream>>>(st, 2048);
    k_f32_to_f16<<<160, 256, 0, stream>>>(w1, w1h, 40960);
    k_f32_to_f16<<<32, 256, 0, stream>>>(w2, w2h, 8192);

    k_conv1_stats<<<NB / 256, 256, 0, stream>>>(x, cw1, cb1, st + 0, st + 8);
    k_finalize<<<1, 8, 0, stream>>>(st + 0, st + 8, gc1, btc1, st + 16, st + 24, 8, fB * 4.0f);

    k_conv2_stats<<<NB / 256, 256, 0, stream>>>(x, cw1, cb1, st + 16, st + 24, cw2, cb2,
                                                st + 32, st + 96);
    k_finalize<<<1, 64, 0, stream>>>(st + 32, st + 96, gc2, btc2, st + 160, st + 224, 64, fB * 6.0f);

    k_front_lstm<<<NB / 256, 256, 0, stream>>>(x, cw1, cb1, st + 16, st + 24, cw2, cb2,
                                               st + 160, st + 224, Wih0, Wih, Whh, bih, bhh, fbuf);

    k_gemm1_wmma<<<dim3(NB / 128, 8), 256, 0, stream>>>(fbuf, w1h, b1, y1);
    k_colstats<<<256, 256, 0, stream>>>(y1, 128, st + 288, st + 416);
    k_finalize<<<1, 128, 0, stream>>>(st + 288, st + 416, g1, bt1, st + 544, st + 672, 128, fB);

    k_gemm2_wmma<<<dim3(NB / 128, 4), 256, 0, stream>>>(y1, st + 544, st + 672, w2h, b2, y2);
    k_colstats<<<256, 256, 0, stream>>>(y2, 64, st + 800, st + 864);
    k_finalize<<<1, 64, 0, stream>>>(st + 800, st + 864, g2, bt2, st + 928, st + 992, 64, fB);

    k_l3<<<NB / 256, 256, 0, stream>>>(y2, st + 928, st + 992, w3, b3, y3, st + 1056, st + 1060);
    k_finalize<<<1, 4, 0, stream>>>(st + 1056, st + 1060, g3, bt3, st + 1064, st + 1068, 4, fB);

    k_l4<<<NB / 256, 256, 0, stream>>>(y3, st + 1064, st + 1068, w4, b4, y4, st + 1072, st + 1104);
    k_finalize<<<1, 32, 0, stream>>>(st + 1072, st + 1104, g4, bt4, st + 1136, st + 1168, 32, fB);

    k_l5<<<NB / 256, 256, 0, stream>>>(y4, st + 1136, st + 1168, w5, b5, y5, st + 1200, st + 1232);
    k_finalize<<<1, 32, 0, stream>>>(st + 1200, st + 1232, g5, bt5, st + 1264, st + 1296, 32, fB);

    k_final<<<NB / 256, 256, 0, stream>>>(y3, st + 1064, st + 1068, y4, st + 1136, st + 1168,
                                          y5, st + 1264, st + 1296, w6, b6, w7, b7, w8, b8, out);
    (void)in_sizes; (void)n_in; (void)out_size; (void)ws_size;
}
